// AttentionBlock_21303037788401
// MI455X (gfx1250) — matrix operand order repaired and checked by reading, not compiled, not run on hardware
//
#include <hip/hip_runtime.h>
#include <math.h>

// ---------------------------------------------------------------------------
// AttentionBlock for MI455X (gfx1250): GroupNorm -> QKV GEMM -> flash attn ->
// proj GEMM + residual.  All matmuls use v_wmma_f32_16x16x32_f16 (wave32).
// Activations are stored t-major by their producers so every GEMM/attention
// LDS tile copy is b128 global load -> b128 LDS store, and every fragment is
// two ds_load_b128.  GEMM mainloop is double-buffered: one barrier per K-step.
// ---------------------------------------------------------------------------

typedef __attribute__((ext_vector_type(16))) _Float16 v16h;
typedef __attribute__((ext_vector_type(8)))  float    v8f;

#define B_   16
#define C_   512
#define T_   1024
#define G_   32
#define H_   8
#define HD_  64
// scale = 1/sqrt(sqrt(64)); applied to q and k rows of qkv (folded: scores/8)
#define QK_SCALE 0.35355339059327379f

#define LDA_  32   // K leading dim of GEMM A tiles
#define LDBT_ 40   // padded K leading dim of GEMM B^T tiles (80B rows, 16B ok)
#define LDATT 72   // padded leading dim for attention tiles (144B rows, 16B ok)
#define LDS_C 65   // padded score-tile leading dim (conflict-free row scan)

// ---- WMMA fragment loaders --------------------------------------------------
// A fragment (16x32 f16) from row-major tile: lane(half,m) reads
// k = half*8 + 0..7  and  k = 16 + half*8 + 0..7   -> two b128 loads.
__device__ __forceinline__ v16h frag_a_rm(const _Float16* tile, int ld) {
  int lane = threadIdx.x & 31;
  int m = lane & 15, half = lane >> 4;
  const _Float16* p = tile + m * ld + half * 8;
  union { v16h v; uint4 u[2]; } r;
  r.u[0] = *(const uint4*)(p);
  r.u[1] = *(const uint4*)(p + 16);
  return r.v;
}

// B fragment (32x16 f16) from N-major tile BsT[n][k]: lane(half,n) reads
// k = half*8 + 0..7  and  k = 16 + half*8 + 0..7   -> two b128 loads.
__device__ __forceinline__ v16h frag_b_nm(const _Float16* tile, int ld) {
  int lane = threadIdx.x & 31;
  int n = lane & 15, half = lane >> 4;
  const _Float16* p = tile + n * ld + half * 8;
  union { v16h v; uint4 u[2]; } r;
  r.u[0] = *(const uint4*)(p);
  r.u[1] = *(const uint4*)(p + 16);
  return r.v;
}

__device__ __forceinline__ v8f wmma_f16(v16h a, v16h b, v8f c) {
  return __builtin_amdgcn_wmma_f32_16x16x32_f16(false, a, false, b, (short)0, c,
                                                false, false);
}

// ---------------------------------------------------------------------------
// Kernel 0: f32 -> f16 convert (weights)
// ---------------------------------------------------------------------------
__global__ void f32_to_f16_kernel(const float* __restrict__ in,
                                  _Float16* __restrict__ out, int n) {
  int i = blockIdx.x * blockDim.x + threadIdx.x;
  if (i < n) out[i] = (_Float16)in[i];
}

// ---------------------------------------------------------------------------
// Kernel 1: GroupNorm.  One block per (b, group): 16 channels x 1024 T.
// xn32 stays c-major (residual path); xnT is t-major [b][t][c] (GEMM operand).
// ---------------------------------------------------------------------------
__global__ void groupnorm_kernel(const float* __restrict__ x,
                                 const float* __restrict__ gamma,
                                 const float* __restrict__ beta,
                                 float* __restrict__ xn32,
                                 _Float16* __restrict__ xnT) {
  const int cpg = C_ / G_;                 // 16
  const int n   = cpg * T_;                // 16384
  int g = blockIdx.x % G_, b = blockIdx.x / G_;
  long base = (long)b * C_ * T_ + (long)g * cpg * T_;

  const float4* x4 = (const float4*)(x + base);
  float s = 0.f, s2 = 0.f;
  for (int i = threadIdx.x; i < n / 4; i += blockDim.x) {
    float4 v = x4[i];
    s  += v.x + v.y + v.z + v.w;
    s2 += v.x * v.x + v.y * v.y + v.z * v.z + v.w * v.w;
  }
  for (int off = 16; off > 0; off >>= 1) {
    s  += __shfl_down(s,  off, 32);
    s2 += __shfl_down(s2, off, 32);
  }
  __shared__ float red0[8], red1[8];
  int wid = threadIdx.x >> 5;
  if ((threadIdx.x & 31) == 0) { red0[wid] = s; red1[wid] = s2; }
  __syncthreads();
  if (threadIdx.x == 0) {
    float a = 0.f, c = 0.f;
    for (int i = 0; i < 8; ++i) { a += red0[i]; c += red1[i]; }
    red0[0] = a; red1[0] = c;
  }
  __syncthreads();
  float mean = red0[0] / (float)n;
  float var  = red1[0] / (float)n - mean * mean;
  float rinv = rsqrtf(var + 1e-5f);

  float4* o4 = (float4*)(xn32 + base);
  for (int i = threadIdx.x; i < n / 4; i += blockDim.x) {
    int local = i * 4;
    int cl = local / T_;           // channel within group
    int t  = local % T_;           // first of 4 consecutive t
    int cg = g * cpg + cl;
    float gm = gamma[cg] * rinv, bt = beta[cg] - mean * gamma[cg] * rinv;
    float4 v = x4[i];
    float4 o;
    o.x = v.x * gm + bt; o.y = v.y * gm + bt;
    o.z = v.z * gm + bt; o.w = v.w * gm + bt;
    o4[i] = o;
    long row = ((long)b * T_ + t) * C_ + cg;   // t-major scatter (write once)
    xnT[row]          = (_Float16)o.x;
    xnT[row + C_]     = (_Float16)o.y;
    xnT[row + 2 * C_] = (_Float16)o.z;
    xnT[row + 3 * C_] = (_Float16)o.w;
  }
}

// ---------------------------------------------------------------------------
// Shared GEMM mainloop: 128(M) x 64(N) tile, BK=32, 8 waves, 2x2 acc/wave.
// A row-major [M,K]; B N-major [N,K] in global (t-major activations).
// Double-buffered LDS: single barrier per K-step.
// ---------------------------------------------------------------------------
__device__ __forceinline__ void gemm_mainloop_128x64(
    const _Float16* __restrict__ Ag, int lda,
    const _Float16* __restrict__ BgT, int ldbT,
    int m0, int n0, int K,
    _Float16 (*As)[128][LDA_], _Float16 (*BsT)[64][LDBT_],
    v8f& a00, v8f& a01, v8f& a10, v8f& a11) {
  int tid = threadIdx.x;
  int w = tid >> 5;
  int mi = (w & 3) * 2;    // wave covers m subtiles mi, mi+1 (of 8)
  int ni = (w >> 2) * 2;   // wave covers n subtiles ni, ni+1 (of 4)
  int am = tid >> 1, ak = (tid & 1) * 16;   // A: 16 halves/thread
  int bn = tid >> 2, bk = (tid & 3) * 8;    // B: 8 halves/thread
  const _Float16* aSrc = Ag + (long)(m0 + am) * lda + ak;
  const _Float16* bSrc = BgT + (long)(n0 + bn) * ldbT + bk;

  auto load_tile = [&](int k0, int buf) {
    *(uint4*)(&As[buf][am][ak])     = *(const uint4*)(aSrc + k0);
    *(uint4*)(&As[buf][am][ak + 8]) = *(const uint4*)(aSrc + k0 + 8);
    *(uint4*)(&BsT[buf][bn][bk])    = *(const uint4*)(bSrc + k0);
  };

  load_tile(0, 0);
  int nIter = K / 32;
  for (int it = 0; it < nIter; ++it) {
    __syncthreads();
    if (it + 1 < nIter) load_tile((it + 1) * 32, (it + 1) & 1);
    int cur = it & 1;
    v16h af0 = frag_a_rm(&As[cur][(mi + 0) * 16][0], LDA_);
    v16h af1 = frag_a_rm(&As[cur][(mi + 1) * 16][0], LDA_);
    v16h bf0 = frag_b_nm(&BsT[cur][(ni + 0) * 16][0], LDBT_);
    v16h bf1 = frag_b_nm(&BsT[cur][(ni + 1) * 16][0], LDBT_);
    a00 = wmma_f16(af0, bf0, a00);
    a01 = wmma_f16(af0, bf1, a01);
    a10 = wmma_f16(af1, bf0, a10);
    a11 = wmma_f16(af1, bf1, a11);
  }
}

// ---------------------------------------------------------------------------
// Kernel 2: QKV GEMM.  q/k rows -> t-major qkT[b][t][0..2C) with bias+scale;
// v rows -> c-major Vbuf[b][c][t] with bias.  Branch is block-uniform.
// ---------------------------------------------------------------------------
__global__ void qkv_gemm_kernel(const _Float16* __restrict__ Wq,   // [3C, C]
                                const _Float16* __restrict__ XnT,  // [B, T, C]
                                const float* __restrict__ bias,    // [3C]
                                _Float16* __restrict__ qkT,        // [B, T, 2C]
                                _Float16* __restrict__ Vbuf) {     // [B, C, T]
  __shared__ __align__(16) _Float16 As[2][128][LDA_];
  __shared__ __align__(16) _Float16 BsT[2][64][LDBT_];
  int b  = blockIdx.z;
  int m0 = blockIdx.y * 128;
  int n0 = blockIdx.x * 64;
  const _Float16* Bbase = XnT + (long)b * T_ * C_;

  v8f a00 = {}, a01 = {}, a10 = {}, a11 = {};
  gemm_mainloop_128x64(Wq, C_, Bbase, C_, m0, n0, C_, As, BsT,
                       a00, a01, a10, a11);

  int tid = threadIdx.x, w = tid >> 5, lane = tid & 31;
  int mi = (w & 3) * 2, ni = (w >> 2) * 2;
  int n = lane & 15, half = lane >> 4;
  _Float16* qkb = qkT + (long)b * T_ * (2 * C_);
  _Float16* vb  = Vbuf + (long)b * C_ * T_;
#pragma unroll
  for (int x = 0; x < 2; ++x) {
    const v8f& c0 = x ? a10 : a00;
    const v8f& c1 = x ? a11 : a01;
#pragma unroll
    for (int r = 0; r < 8; ++r) {
      int o = m0 + (mi + x) * 16 + r + 8 * half;
      float bv = bias[o];
      int tt = n0 + ni * 16 + n;
      if (o < 2 * C_) {   // q/k: scale, t-major scatter (write once)
        qkb[(long)tt * (2 * C_) + o]        = (_Float16)((c0[r] + bv) * QK_SCALE);
        qkb[(long)(tt + 16) * (2 * C_) + o] = (_Float16)((c1[r] + bv) * QK_SCALE);
      } else {            // v: c-major, lane-contiguous runs
        long vrow = (long)(o - 2 * C_) * T_;
        vb[vrow + tt]      = (_Float16)(c0[r] + bv);
        vb[vrow + tt + 16] = (_Float16)(c1[r] + bv);
      }
    }
  }
}

// ---------------------------------------------------------------------------
// Kernel 3: flash attention, one block per (bh, 64 query rows).
// Q tile [t][c] and K tile [s][c] come straight from t-major qkT (b128 copies);
// V tile [c][s] straight from c-major Vbuf.  Output h is stored t-major.
// ---------------------------------------------------------------------------
__global__ void attn_kernel(const _Float16* __restrict__ qkT,   // [B, T, 2C]
                            const _Float16* __restrict__ Vbuf,  // [B, C, T]
                            _Float16* __restrict__ HT) {        // [B, T, C]
  __shared__ __align__(16) _Float16 Qt[64][LDATT];   // [t][c]  A operand
  __shared__ __align__(16) _Float16 KtN[64][LDATT];  // [s][c]  B: n=s, k=c
  __shared__ __align__(16) _Float16 VtN[64][LDATT];  // [c][s]  B: n=c, k=s
  __shared__ __align__(16) _Float16 Pt[64][LDATT];   // [t][s]  A operand
  __shared__ float Sc[64][LDS_C];
  __shared__ float mrow[64], lrow[64], arow[64];

  int bh = blockIdx.y;
  int b = bh >> 3, h = bh & 7;
  int t0 = blockIdx.x * 64;
  const _Float16* qkb = qkT + (long)b * T_ * (2 * C_);
  const _Float16* vb  = Vbuf + (long)b * C_ * T_ + (long)(h * HD_) * T_;

  int tid = threadIdx.x;
  int w = tid >> 5, lane = tid & 31;
  int ti = w & 3, nh = w >> 2;
  int n = lane & 15, half = lane >> 4;
  int rr = tid >> 2, ro = (tid & 3) * 16;   // cooperative copy coords

  {  // Q tile: rows t0+rr, cols h*64 + ro..ro+15 (two b128 each)
    const _Float16* src = qkb + (long)(t0 + rr) * (2 * C_) + h * HD_ + ro;
    *(uint4*)(&Qt[rr][ro])     = *(const uint4*)(src);
    *(uint4*)(&Qt[rr][ro + 8]) = *(const uint4*)(src + 8);
  }
  if (tid < 64) { mrow[tid] = -1e30f; lrow[tid] = 0.f; }
  v8f o0 = {}, o1 = {};
  __syncthreads();

  for (int s0 = 0; s0 < T_; s0 += 64) {
    {  // K tile: rows s0+rr, cols C + h*64 + ro
      const _Float16* src = qkb + (long)(s0 + rr) * (2 * C_) + C_ + h * HD_ + ro;
      *(uint4*)(&KtN[rr][ro])     = *(const uint4*)(src);
      *(uint4*)(&KtN[rr][ro + 8]) = *(const uint4*)(src + 8);
    }
    {  // V tile: rows c=rr, cols s0+ro
      const _Float16* src = vb + (long)rr * T_ + s0 + ro;
      *(uint4*)(&VtN[rr][ro])     = *(const uint4*)(src);
      *(uint4*)(&VtN[rr][ro + 8]) = *(const uint4*)(src + 8);
    }
    __syncthreads();

    // S = q^T k  (K dim = 64 channels)
    v8f sA0 = {}, sA1 = {};
#pragma unroll
    for (int kk = 0; kk < 64; kk += 32) {
      v16h a   = frag_a_rm(&Qt[ti * 16][kk], LDATT);
      v16h bb0 = frag_b_nm(&KtN[(nh * 2 + 0) * 16][kk], LDATT);
      v16h bb1 = frag_b_nm(&KtN[(nh * 2 + 1) * 16][kk], LDATT);
      sA0 = wmma_f16(a, bb0, sA0);
      sA1 = wmma_f16(a, bb1, sA1);
    }
#pragma unroll
    for (int r = 0; r < 8; ++r) {
      int m = ti * 16 + r + 8 * half;
      Sc[m][(nh * 2) * 16 + n]     = sA0[r];
      Sc[m][(nh * 2 + 1) * 16 + n] = sA1[r];
    }
    __syncthreads();

    // online softmax per query row (padded LDS -> conflict-free row scan)
    if (tid < 64) {
      float mo = mrow[tid];
      float rm = -1e30f;
      for (int s = 0; s < 64; ++s) rm = fmaxf(rm, Sc[tid][s]);
      float nm = fmaxf(mo, rm);
      float al = __expf(mo - nm);
      float sum = 0.f;
      for (int s = 0; s < 64; ++s) {
        float e = __expf(Sc[tid][s] - nm);
        Pt[tid][s] = (_Float16)e;
        sum += e;
      }
      mrow[tid] = nm;
      lrow[tid] = lrow[tid] * al + sum;
      arow[tid] = al;
    }
    __syncthreads();

    // rescale O accumulators, then O += P * v^T
#pragma unroll
    for (int r = 0; r < 8; ++r) {
      float al = arow[ti * 16 + r + 8 * half];
      o0[r] *= al; o1[r] *= al;
    }
#pragma unroll
    for (int kk = 0; kk < 64; kk += 32) {
      v16h a   = frag_a_rm(&Pt[ti * 16][kk], LDATT);
      v16h bb0 = frag_b_nm(&VtN[(nh * 2 + 0) * 16][kk], LDATT);
      v16h bb1 = frag_b_nm(&VtN[(nh * 2 + 1) * 16][kk], LDATT);
      o0 = wmma_f16(a, bb0, o0);
      o1 = wmma_f16(a, bb1, o1);
    }
    __syncthreads();
  }

  // HT[b][t0 + t][h*64 + c] = O[t][c] / l[t]  (lane-contiguous 32B runs)
  _Float16* ob = HT + (long)b * T_ * C_ + h * HD_;
#pragma unroll
  for (int r = 0; r < 8; ++r) {
    int m = ti * 16 + r + 8 * half;        // local t
    float inv = 1.0f / lrow[m];
    long row = (long)(t0 + m) * C_;
    int c0 = (nh * 2) * 16 + n;
    ob[row + c0]      = (_Float16)(o0[r] * inv);
    ob[row + c0 + 16] = (_Float16)(o1[r] * inv);
  }
}

// ---------------------------------------------------------------------------
// Kernel 4: proj GEMM + bias + f32 residual -> d_out
// ---------------------------------------------------------------------------
__global__ void proj_gemm_kernel(const _Float16* __restrict__ Wp,   // [C, C]
                                 const _Float16* __restrict__ HT,   // [B, T, C]
                                 const float* __restrict__ bias,    // [C]
                                 const float* __restrict__ Xn32,    // [B, C, T]
                                 float* __restrict__ Out) {         // [B, C, T]
  __shared__ __align__(16) _Float16 As[2][128][LDA_];
  __shared__ __align__(16) _Float16 BsT[2][64][LDBT_];
  int b  = blockIdx.z;
  int m0 = blockIdx.y * 128;
  int n0 = blockIdx.x * 64;
  const _Float16* Bbase = HT + (long)b * T_ * C_;

  v8f a00 = {}, a01 = {}, a10 = {}, a11 = {};
  gemm_mainloop_128x64(Wp, C_, Bbase, C_, m0, n0, C_, As, BsT,
                       a00, a01, a10, a11);

  int tid = threadIdx.x, w = tid >> 5, lane = tid & 31;
  int mi = (w & 3) * 2, ni = (w >> 2) * 2;
  int n = lane & 15, half = lane >> 4;
  const float* rbase = Xn32 + (long)b * C_ * T_;
  float* obase = Out + (long)b * C_ * T_;
#pragma unroll
  for (int x = 0; x < 2; ++x) {
    const v8f& c0 = x ? a10 : a00;
    const v8f& c1 = x ? a11 : a01;
#pragma unroll
    for (int r = 0; r < 8; ++r) {
      int o = m0 + (mi + x) * 16 + r + 8 * half;
      float bv = bias[o];
      int tt = n0 + ni * 16 + n;
      obase[(long)o * T_ + tt]      = c0[r] + bv + rbase[(long)o * T_ + tt];
      obase[(long)o * T_ + tt + 16] = c1[r] + bv + rbase[(long)o * T_ + tt + 16];
    }
  }
}

// ---------------------------------------------------------------------------
extern "C" void kernel_launch(void* const* d_in, const int* in_sizes, int n_in,
                              void* d_out, int out_size, void* d_ws,
                              size_t ws_size, hipStream_t stream) {
  const float* x      = (const float*)d_in[0];
  const float* gamma  = (const float*)d_in[1];
  const float* beta   = (const float*)d_in[2];
  const float* w_qkv  = (const float*)d_in[3];
  const float* b_qkv  = (const float*)d_in[4];
  const float* w_proj = (const float*)d_in[5];
  const float* b_proj = (const float*)d_in[6];
  float* out = (float*)d_out;

  char* ws = (char*)d_ws;
  size_t off = 0;
  auto wsalloc = [&](size_t bytes) -> char* {
    char* p = ws + off;
    off += (bytes + 255) & ~(size_t)255;
    return p;
  };
  float*    xn32 = (float*)   wsalloc((size_t)B_ * C_ * T_ * 4);
  _Float16* xnT  = (_Float16*)wsalloc((size_t)B_ * T_ * C_ * 2);
  _Float16* wq16 = (_Float16*)wsalloc((size_t)3 * C_ * C_ * 2);
  _Float16* wp16 = (_Float16*)wsalloc((size_t)C_ * C_ * 2);
  _Float16* qkT  = (_Float16*)wsalloc((size_t)B_ * T_ * 2 * C_ * 2);
  _Float16* vbuf = (_Float16*)wsalloc((size_t)B_ * C_ * T_ * 2);
  _Float16* hT   = (_Float16*)wsalloc((size_t)B_ * T_ * C_ * 2);

  {
    int nq = 3 * C_ * C_;
    f32_to_f16_kernel<<<(nq + 255) / 256, 256, 0, stream>>>(w_qkv, wq16, nq);
    int np = C_ * C_;
    f32_to_f16_kernel<<<(np + 255) / 256, 256, 0, stream>>>(w_proj, wp16, np);
  }
  groupnorm_kernel<<<B_ * G_, 256, 0, stream>>>(x, gamma, beta, xn32, xnT);
  qkv_gemm_kernel<<<dim3(T_ / 64, (3 * C_) / 128, B_), 256, 0, stream>>>(
      wq16, xnT, b_qkv, qkT, vbuf);
  attn_kernel<<<dim3(T_ / 64, B_ * H_), 256, 0, stream>>>(qkT, vbuf, hT);
  proj_gemm_kernel<<<dim3(T_ / 64, C_ / 128, B_), 256, 0, stream>>>(
      wp16, hT, b_proj, xn32, out);
}
